// TemplateEncoder_34084860461439
// MI455X (gfx1250) — matrix operand order repaired and checked by reading, not compiled, not run on hardware
//
#include <hip/hip_runtime.h>
#include <hip/hip_bf16.h>

typedef __attribute__((ext_vector_type(16))) _Float16 v16h;
typedef __attribute__((ext_vector_type(8)))  _Float16 v8h;
typedef __attribute__((ext_vector_type(8)))  float    v8f;

// Problem constants (from setup_inputs)
constexpr int kB = 2, kN = 32, kT = 32, kM = 256, kL = 20;
constexpr int kH = 128, kD = 128, kTAU = 64;
constexpr int kBM = kB * kM;            // 512 polylines
constexpr int kRowsPoly = kBM * kL;     // 10240 polyline points
constexpr int kRowsAgt  = kB * kN * kT; // 2048 agent-time rows
constexpr float kInvSqrtD = 0.08838834764831845f; // 1/sqrt(128)
constexpr float kRadius = 30.0f;

// ---------------------------------------------------------------------------
// Weight pre-pack: WT[n*K + k] = (f16) W[k*Kout + n]   (transpose + convert)
// Makes every B-fragment lane read two contiguous 16-byte runs.
// ---------------------------------------------------------------------------
__launch_bounds__(256)
__global__ void pack_wT_kernel(const float* __restrict__ W, _Float16* __restrict__ WT,
                               int K, int Kout)
{
  int idx = blockIdx.x * blockDim.x + threadIdx.x;
  if (idx >= K * Kout) return;
  int k = idx / Kout, n = idx - k * Kout;
  WT[(long)n * K + k] = (_Float16)W[idx];
}

// ---------------------------------------------------------------------------
// WMMA strip GEMM: one wave per 16-row strip, NT column tiles (Kout = NT*16).
// Requirements (guaranteed by caller): rows%16==0, K%32==0, A base & lda 16B-aligned.
// out[row,col] = act(A[row,:K] @ W[:K,col] + bias[col]) * mask(row)
// ---------------------------------------------------------------------------
template<int ACT, int NT>
__launch_bounds__(32)
__global__ void gemm_strip_kernel(const float* __restrict__ A, int lda,
                                  const _Float16* __restrict__ WT, // (NT*16) x K f16
                                  const float* __restrict__ bias,
                                  const float* __restrict__ row_mask, // rows (or null)
                                  float* __restrict__ out, int ldout,
                                  int K)
{
  const int tileM = blockIdx.x << 4;
  const int lane  = threadIdx.x;      // 0..31 (wave32)
  const int half  = lane >> 4;        // half-wave select
  const int lr    = lane & 15;

  v8f acc[NT];
#pragma unroll
  for (int nt = 0; nt < NT; ++nt) acc[nt] = (v8f){};

  const float* __restrict__ Arow = A + (long)(tileM + lr) * lda;

  for (int k0 = 0; k0 < K; k0 += 32) {
    // A 16x32 f16 fragment: lanes 0-15 K={0..7,16..23}, lanes 16-31 K={8..15,24..31}
    const float4 a0 = *(const float4*)(Arow + k0 + 8 * half);
    const float4 a1 = *(const float4*)(Arow + k0 + 8 * half + 4);
    const float4 a2 = *(const float4*)(Arow + k0 + 16 + 8 * half);
    const float4 a3 = *(const float4*)(Arow + k0 + 16 + 8 * half + 4);
    v16h a;
    a[0]  = (_Float16)a0.x; a[1]  = (_Float16)a0.y; a[2]  = (_Float16)a0.z; a[3]  = (_Float16)a0.w;
    a[4]  = (_Float16)a1.x; a[5]  = (_Float16)a1.y; a[6]  = (_Float16)a1.z; a[7]  = (_Float16)a1.w;
    a[8]  = (_Float16)a2.x; a[9]  = (_Float16)a2.y; a[10] = (_Float16)a2.z; a[11] = (_Float16)a2.w;
    a[12] = (_Float16)a3.x; a[13] = (_Float16)a3.y; a[14] = (_Float16)a3.z; a[15] = (_Float16)a3.w;

#pragma unroll
    for (int nt = 0; nt < NT; ++nt) {
      // B 32x16 f16 fragment: half-wave h covers K = {8h..8h+7, 16+8h..16+8h+7}, col = lr
      const _Float16* __restrict__ wrow = WT + (long)(nt * 16 + lr) * K + k0 + 8 * half;
      const v8h b0 = *(const v8h*)(wrow);
      const v8h b1 = *(const v8h*)(wrow + 16);
      v16h b;
#pragma unroll
      for (int e = 0; e < 8; ++e) { b[e] = b0[e]; b[8 + e] = b1[e]; }
      acc[nt] = __builtin_amdgcn_wmma_f32_16x16x32_f16(
          false, a, false, b, (short)0, acc[nt], false, false);
    }
  }

  float rmask[8];
#pragma unroll
  for (int r = 0; r < 8; ++r)
    rmask[r] = row_mask ? ((row_mask[tileM + half * 8 + r] > 0.5f) ? 1.0f : 0.0f) : 1.0f;

#pragma unroll
  for (int nt = 0; nt < NT; ++nt) {
    const int col = nt * 16 + lr;
    const float bv = bias[col];
#pragma unroll
    for (int r = 0; r < 8; ++r) {        // C/D layout: reg r -> row r + 8*half
      const int row = tileM + half * 8 + r;
      float v = acc[nt][r] + bv;
      if (ACT) v = fmaxf(v, 0.0f);
      v *= rmask[r];
      out[(long)row * ldout + col] = v;
    }
  }
}

// ---------------------------------------------------------------------------
// Tiny-K first layers (K=2 / K=5): pure VALU, one thread per output element.
// ---------------------------------------------------------------------------
__launch_bounds__(256)
__global__ void poly_l1_kernel(const float* __restrict__ poly,  // rows x 2
                               const float* __restrict__ w1,    // 2 x 128
                               const float* __restrict__ b1,
                               float* __restrict__ h1)          // rows x 128
{
  int idx = blockIdx.x * blockDim.x + threadIdx.x;
  if (idx >= kRowsPoly * kH) return;
  int row = idx >> 7, h = idx & 127;
  float x0 = poly[row * 2 + 0], x1 = poly[row * 2 + 1];
  h1[idx] = fmaxf(x0 * w1[h] + x1 * w1[kH + h] + b1[h], 0.0f);
}

__launch_bounds__(256)
__global__ void agent_l1_kernel(const float* __restrict__ st,   // rows x 5
                                const float* __restrict__ w1,   // 5 x 128
                                const float* __restrict__ b1,
                                float* __restrict__ h1)         // rows x 128
{
  int idx = blockIdx.x * blockDim.x + threadIdx.x;
  if (idx >= kRowsAgt * kH) return;
  int row = idx >> 7, h = idx & 127;
  const float* x = st + (long)row * 5;
  float v = b1[h];
#pragma unroll
  for (int f = 0; f < 5; ++f) v += x[f] * w1[f * kH + h];
  h1[idx] = fmaxf(v, 0.0f);
}

// ---------------------------------------------------------------------------
// Max-pool over L points + type/tl/route embedding add.
// ---------------------------------------------------------------------------
__launch_bounds__(256)
__global__ void poly_max_emb_kernel(const float* __restrict__ h2,   // (BM*L) x 128
                                    const int* __restrict__ ptype,
                                    const int* __restrict__ tl,
                                    const int* __restrict__ route,
                                    const float* __restrict__ type_emb,
                                    const float* __restrict__ tl_emb,
                                    const float* __restrict__ route_emb,
                                    float* __restrict__ g)           // BM x 128
{
  int idx = blockIdx.x * blockDim.x + threadIdx.x;
  if (idx >= kBM * kH) return;
  int bm = idx >> 7, h = idx & 127;
  float mx = -3.4e38f;
#pragma unroll
  for (int l = 0; l < kL; ++l)
    mx = fmaxf(mx, h2[((long)bm * kL + l) * kH + h]);
  int ti = min(max(ptype[bm], 0), 3);
  int si = min(max(tl[bm], 0), 7);
  int ri = min(max(route[bm], 0), 1);
  g[idx] = mx + type_emb[ti * kH + h] + tl_emb[si * kH + h] + route_emb[ri * kH + h];
}

__launch_bounds__(256)
__global__ void map_center_kernel(const float* __restrict__ poly, float* __restrict__ center)
{
  int bm = blockIdx.x * blockDim.x + threadIdx.x;
  if (bm >= kBM) return;
  float sx = 0.f, sy = 0.f;
#pragma unroll
  for (int l = 0; l < kL; ++l) {
    sx += poly[((long)bm * kL + l) * 2 + 0];
    sy += poly[((long)bm * kL + l) * 2 + 1];
  }
  center[bm * 2 + 0] = sx * (1.0f / kL);
  center[bm * 2 + 1] = sy * (1.0f / kL);
}

// ---------------------------------------------------------------------------
// Agent->map cross attention: one block per (b,n,t) query; 256 threads = 256 keys.
// ---------------------------------------------------------------------------
__launch_bounds__(256)
__global__ void map_attn_kernel(float* __restrict__ tau,              // 2048 x 384
                                const float* __restrict__ agents_state,
                                const float* __restrict__ map_center, // 512 x 2
                                const float* __restrict__ map_node,   // 512 x 128
                                const float* __restrict__ map_poly_mask,
                                const float* __restrict__ mr_w1,
                                const float* __restrict__ mr_b1,
                                const float* __restrict__ mr_w2,
                                const float* __restrict__ mr_b2)
{
  __shared__ float qs[kD];
  __shared__ float attn[kM];
  __shared__ float red[kM];

  const int q   = blockIdx.x;      // 0..2047
  const int tid = threadIdx.x;     // 0..255
  const int b   = q / (kN * kT);
  const long arow = q;

  if (tid < kD) qs[tid] = tau[arow * 384 + tid];
  __syncthreads();

  const float ax = agents_state[arow * 5 + 0];
  const float ay = agents_state[arow * 5 + 1];

  const int gm = b * kM + tid;
  const float cx = map_center[gm * 2 + 0] - ax;
  const float cy = map_center[gm * 2 + 1] - ay;
  const float dist = sqrtf(cx * cx + cy * cy);

  float s = mr_b2[0];
  for (int h = 0; h < kH; ++h) {
    float hv = cx * mr_w1[h] + cy * mr_w1[kH + h] + dist * mr_w1[2 * kH + h] + mr_b1[h];
    s += fmaxf(hv, 0.0f) * mr_w2[h];
  }
  float dot = 0.0f;
  for (int d = 0; d < kD; ++d) dot += qs[d] * map_node[(long)gm * kD + d];
  dot *= kInvSqrtD;

  const float mk = (map_poly_mask[gm] > 0.5f) ? 1.0f : 0.0f;
  const float logit = (mk > 0.0f) ? (dot + s) : -1.0e9f;

  red[tid] = logit; __syncthreads();
  for (int st = 128; st > 0; st >>= 1) { if (tid < st) red[tid] = fmaxf(red[tid], red[tid + st]); __syncthreads(); }
  const float mx = red[0]; __syncthreads();
  const float e = __expf(logit - mx) * mk;
  red[tid] = e; __syncthreads();
  for (int st = 128; st > 0; st >>= 1) { if (tid < st) red[tid] += red[tid + st]; __syncthreads(); }
  const float inv = 1.0f / fmaxf(red[0], 1e-9f);
  attn[tid] = e * inv;
  __syncthreads();

  for (int d = tid; d < kD; d += 256) {
    float acc = 0.0f;
    for (int mm = 0; mm < kM; ++mm)
      acc += attn[mm] * map_node[(long)(b * kM + mm) * kD + d];
    tau[arow * 384 + kD + d] = acc;
  }
}

// ---------------------------------------------------------------------------
// Agent->agent attention per (b,t): 32 agents resident in LDS.
// ---------------------------------------------------------------------------
__launch_bounds__(256)
__global__ void nbr_attn_kernel(float* __restrict__ tau,
                                const float* __restrict__ agents_state,
                                const float* __restrict__ agents_mask,
                                const float* __restrict__ nr_w1,
                                const float* __restrict__ nr_b1,
                                const float* __restrict__ nr_w2,
                                const float* __restrict__ nr_b2)
{
  __shared__ float As[kN * kD];     // 16 KB
  __shared__ float Ls[kN * kN];
  __shared__ float Vs[kN * kN];
  __shared__ float px[kN], py[kN], vx[kN], vy[kN], mk[kN];

  const int bt = blockIdx.x;        // 0..B*T-1
  const int b = bt / kT, t = bt % kT;
  const int tid = threadIdx.x;

  for (int idx = tid; idx < kN * kD; idx += 256) {
    int i = idx >> 7, d = idx & 127;
    long row = ((long)(b * kN + i)) * kT + t;
    As[idx] = tau[row * 384 + d];
  }
  if (tid < kN) {
    int i = tid;
    long row = ((long)(b * kN + i)) * kT + t;
    px[i] = agents_state[row * 5 + 0];
    py[i] = agents_state[row * 5 + 1];
    vx[i] = agents_state[row * 5 + 3];
    vy[i] = agents_state[row * 5 + 4];
    mk[i] = agents_mask[row];
  }
  __syncthreads();

  for (int p = tid; p < kN * kN; p += 256) {
    int i = p >> 5, j = p & 31;
    float rpx = px[j] - px[i], rpy = py[j] - py[i];
    float rvx = vx[j] - vx[i], rvy = vy[j] - vy[i];
    float d = sqrtf(rpx * rpx + rpy * rpy);
    float s = nr_b2[0];
    for (int h = 0; h < kH; ++h) {
      float hv = rpx * nr_w1[h] + rpy * nr_w1[kH + h] + rvx * nr_w1[2 * kH + h] +
                 rvy * nr_w1[3 * kH + h] + d * nr_w1[4 * kH + h] + nr_b1[h];
      s += fmaxf(hv, 0.0f) * nr_w2[h];
    }
    float dotn = 0.0f;
    for (int dd = 0; dd < kD; ++dd) dotn += As[i * kD + dd] * As[j * kD + dd];
    dotn *= kInvSqrtD;
    bool valid = (mk[i] > 0.5f) && (mk[j] > 0.5f) && (d <= kRadius) && (i != j);
    Ls[p] = valid ? (dotn + s) : -1.0e9f;
    Vs[p] = valid ? 1.0f : 0.0f;
  }
  __syncthreads();

  if (tid < kN) {
    int i = tid;
    float mx = -3.4e38f;
    for (int j = 0; j < kN; ++j) mx = fmaxf(mx, Ls[i * kN + j]);
    float sum = 0.0f;
    for (int j = 0; j < kN; ++j) {
      float e = __expf(Ls[i * kN + j] - mx) * Vs[i * kN + j];
      Ls[i * kN + j] = e;
      sum += e;
    }
    float inv = 1.0f / fmaxf(sum, 1e-9f);
    for (int j = 0; j < kN; ++j) Ls[i * kN + j] *= inv;
  }
  __syncthreads();

  for (int qd = tid; qd < kN * kD; qd += 256) {
    int i = qd >> 7, d = qd & 127;
    float acc = 0.0f;
    for (int j = 0; j < kN; ++j) acc += Ls[i * kN + j] * As[j * kD + d];
    long row = ((long)(b * kN + i)) * kT + t;
    tau[row * 384 + 2 * kD + d] = acc;
  }
}

// ---------------------------------------------------------------------------
extern "C" void kernel_launch(void* const* d_in, const int* in_sizes, int n_in,
                              void* d_out, int out_size, void* d_ws, size_t ws_size,
                              hipStream_t stream)
{
  (void)in_sizes; (void)n_in; (void)out_size; (void)ws_size;
  const float* agents_state  = (const float*)d_in[0];
  const float* agents_mask   = (const float*)d_in[1];
  const float* map_polylines = (const float*)d_in[2];
  const float* map_poly_mask = (const float*)d_in[3];
  const int*   map_poly_type = (const int*)d_in[4];
  const int*   map_tl_status = (const int*)d_in[5];
  const int*   map_on_route  = (const int*)d_in[6];
  const float* pm_w1 = (const float*)d_in[7];
  const float* pm_b1 = (const float*)d_in[8];
  const float* pm_w2 = (const float*)d_in[9];
  const float* pm_b2 = (const float*)d_in[10];
  const float* type_emb  = (const float*)d_in[11];
  const float* tl_emb    = (const float*)d_in[12];
  const float* route_emb = (const float*)d_in[13];
  const float* mo_w1 = (const float*)d_in[14];
  const float* mo_b1 = (const float*)d_in[15];
  const float* mo_w2 = (const float*)d_in[16];
  const float* mo_b2 = (const float*)d_in[17];
  const float* ae_w1 = (const float*)d_in[18];
  const float* ae_b1 = (const float*)d_in[19];
  const float* ae_w2 = (const float*)d_in[20];
  const float* ae_b2 = (const float*)d_in[21];
  const float* ae_w3 = (const float*)d_in[22];
  const float* ae_b3 = (const float*)d_in[23];
  const float* mr_w1 = (const float*)d_in[24];
  const float* mr_b1 = (const float*)d_in[25];
  const float* mr_w2 = (const float*)d_in[26];
  const float* mr_b2 = (const float*)d_in[27];
  const float* nr_w1 = (const float*)d_in[28];
  const float* nr_b1 = (const float*)d_in[29];
  const float* nr_w2 = (const float*)d_in[30];
  const float* nr_b2 = (const float*)d_in[31];
  const float* to_w1 = (const float*)d_in[32];
  const float* to_b1 = (const float*)d_in[33];
  const float* to_w2 = (const float*)d_in[34];
  const float* to_b2 = (const float*)d_in[35];
  float* out = (float*)d_out;

  // ---- workspace layout (fp32 part ~3.61M floats, then f16 packed weights) ----
  float* ws   = (float*)d_ws;
  float* h1   = ws;                                   // 10240 x 128 (reused 2048 x 128)
  float* h2   = h1 + (size_t)kRowsPoly * kH;          // 10240 x 128
  float* g    = h2 + (size_t)kRowsPoly * kH;          // 512 x 128
  float* g2   = g  + (size_t)kBM * kH;                // 512 x 128
  float* mapn = g2 + (size_t)kBM * kH;                // 512 x 128 (map_node)
  float* mctr = mapn + (size_t)kBM * kD;              // 512 x 2
  float* tau  = mctr + (size_t)kBM * 2;               // 2048 x 384
  _Float16* wt = (_Float16*)(tau + (size_t)kRowsAgt * 384);
  _Float16* wt_pm2 = wt;                              // 128 x 128
  _Float16* wt_mo1 = wt_pm2 + 128 * 128;
  _Float16* wt_mo2 = wt_mo1 + 128 * 128;
  _Float16* wt_ae2 = wt_mo2 + 128 * 128;
  _Float16* wt_ae3 = wt_ae2 + 128 * 128;
  _Float16* wt_to1 = wt_ae3 + 128 * 128;              // 128 x 384
  _Float16* wt_to2 = wt_to1 + 128 * 384;              // 64 x 128

  // ---- pack weights (transpose + f16) ----
  auto pgrid = [](int n) { return dim3((n + 255) / 256); };
  pack_wT_kernel<<<pgrid(128 * 128), 256, 0, stream>>>(pm_w2, wt_pm2, 128, 128);
  pack_wT_kernel<<<pgrid(128 * 128), 256, 0, stream>>>(mo_w1, wt_mo1, 128, 128);
  pack_wT_kernel<<<pgrid(128 * 128), 256, 0, stream>>>(mo_w2, wt_mo2, 128, 128);
  pack_wT_kernel<<<pgrid(128 * 128), 256, 0, stream>>>(ae_w2, wt_ae2, 128, 128);
  pack_wT_kernel<<<pgrid(128 * 128), 256, 0, stream>>>(ae_w3, wt_ae3, 128, 128);
  pack_wT_kernel<<<pgrid(384 * 128), 256, 0, stream>>>(to_w1, wt_to1, 384, 128);
  pack_wT_kernel<<<pgrid(128 * 64),  256, 0, stream>>>(to_w2, wt_to2, 128, 64);

  // ---- map polyline encoder ----
  poly_l1_kernel<<<pgrid(kRowsPoly * kH), 256, 0, stream>>>(map_polylines, pm_w1, pm_b1, h1);
  gemm_strip_kernel<1, 8><<<kRowsPoly / 16, 32, 0, stream>>>(
      h1, kH, wt_pm2, pm_b2, nullptr, h2, kH, kH);
  poly_max_emb_kernel<<<pgrid(kBM * kH), 256, 0, stream>>>(
      h2, map_poly_type, map_tl_status, map_on_route, type_emb, tl_emb, route_emb, g);
  map_center_kernel<<<pgrid(kBM), 256, 0, stream>>>(map_polylines, mctr);
  gemm_strip_kernel<1, 8><<<kBM / 16, 32, 0, stream>>>(
      g, kH, wt_mo1, mo_b1, nullptr, g2, kH, kH);
  gemm_strip_kernel<0, 8><<<kBM / 16, 32, 0, stream>>>(
      g2, kH, wt_mo2, mo_b2, map_poly_mask, mapn, kD, kH);

  // ---- agent encoder (h1/h2 reused as 2048 x 128 scratch) ----
  agent_l1_kernel<<<pgrid(kRowsAgt * kH), 256, 0, stream>>>(agents_state, ae_w1, ae_b1, h1);
  gemm_strip_kernel<1, 8><<<kRowsAgt / 16, 32, 0, stream>>>(
      h1, kH, wt_ae2, ae_b2, nullptr, h2, kH, kH);
  gemm_strip_kernel<0, 8><<<kRowsAgt / 16, 32, 0, stream>>>(
      h2, kH, wt_ae3, ae_b3, agents_mask, tau, 384, kH);

  // ---- fused attentions (write tau cols [128,256) and [256,384)) ----
  map_attn_kernel<<<kRowsAgt, 256, 0, stream>>>(
      tau, agents_state, mctr, mapn, map_poly_mask, mr_w1, mr_b1, mr_w2, mr_b2);
  nbr_attn_kernel<<<kB * kT, 256, 0, stream>>>(
      tau, agents_state, agents_mask, nr_w1, nr_b1, nr_w2, nr_b2);

  // ---- output MLP ----
  gemm_strip_kernel<1, 8><<<kRowsAgt / 16, 32, 0, stream>>>(
      tau, 3 * kD, wt_to1, to_b1, nullptr, h1, kH, 3 * kD);
  gemm_strip_kernel<0, 4><<<kRowsAgt / 16, 32, 0, stream>>>(
      h1, kH, wt_to2, to_b2, agents_mask, out, kTAU, kH);
}
